// NetVLAD_47699906789864
// MI455X (gfx1250) — matrix operand order repaired and checked by reading, not compiled, not run on hardware
//
#include <hip/hip_runtime.h>
#include <math.h>

// ---------------- problem constants ----------------
#define NN 16
#define CC 1024
#define PP 1024
#define KK 32
#define PSPLIT 8        // workgroups per image
#define PT 128          // pixels per workgroup
#define TILE_P 32       // pixels per inner tile (WMMA K for vlad GEMM)

typedef __attribute__((ext_vector_type(16))) __bf16 v16bf;
typedef __attribute__((ext_vector_type(8)))  float  v8f;

// ---------------- LDS layout (bytes) ----------------
// conv_w bf16 [32 clusters][1032] (pad -> per-lane b128 loads tile the banks)
#define CW_STRIDE   1032                             // bf16 per row
#define CW_OFF      0
#define CW_BYTES    (KK * CW_STRIDE * 2)             // 66048
// xs1t: logits-GEMM B layout: bf16 [32 pixels][1032 c] -> lane's 16 B elements contiguous
#define XS1_STRIDE  1032                             // bf16 per row (1024 + pad)
#define XS1_OFF     (CW_OFF + CW_BYTES)
#define XS1_BYTES   (TILE_P * XS1_STRIDE * 2)        // 66048
// xs2t: vlad-GEMM B layout: bf16 [1024 c][40 pixels] -> lane's 16 B elements contiguous
#define XS2_STRIDE  40                               // bf16 per row (32 + pad)
#define XS2_OFF     (XS1_OFF + XS1_BYTES)
#define XS2_BYTES   (CC * XS2_STRIDE * 2)            // 81920
// logits / softmax scratch f32 [32 clusters][32 pixels]
#define LG_OFF      (XS2_OFF + XS2_BYTES)
#define LG_BYTES    (KK * TILE_P * 4)                // 4096
// a (assignment) bf16 in A-matrix layout [32 clusters][40 pixels]
#define AL_STRIDE   40
#define AL_OFF      (LG_OFF + LG_BYTES)
#define AL_BYTES    (KK * AL_STRIDE * 2)             // 2560
// per-pixel sumsq partials [8][32] f32
#define RED_OFF     (AL_OFF + AL_BYTES)
#define RED_BYTES   (8 * 32 * 4)
// per-pixel inverse norms [32] f32
#define SC_OFF      (RED_OFF + RED_BYTES)
#define SC_BYTES    (32 * 4)
#define LDS_TOTAL   (SC_OFF + SC_BYTES)              // 221824 B < 320 KB

// workspace layout (floats)
#define WS_VLAD     0                                 // [16][32][1024]
#define WS_ASUM     (NN * KK * CC)                    // [16][32]
#define WS_ROWSS    (WS_ASUM + NN * KK)               // [16][32]
#define WS_FLOATS   (WS_ROWSS + NN * KK)

// ---------------- helper: zero workspace ----------------
__global__ void netvlad_zero_ws(float* __restrict__ ws, int n) {
    int i = blockIdx.x * blockDim.x + threadIdx.x;
    if (i < n) ws[i] = 0.0f;
}

// ---------------- fused main kernel ----------------
// grid = NN*PSPLIT blocks of 256 threads (8 wave32 waves)
__global__ void __launch_bounds__(256)
netvlad_main(const float* __restrict__ x,        // (N,C,H,W) = (16,1024,1024)
             const float* __restrict__ conv_w,   // (K,C) = (32,1024)
             float* __restrict__ vlad_ws,        // (N,K,C) f32 accum
             float* __restrict__ asum_ws) {      // (N,K) f32 accum
    extern __shared__ char smem[];
    __bf16*   cw   = (__bf16*)(smem + CW_OFF);
    __bf16*   xs1  = (__bf16*)(smem + XS1_OFF);  // [pixel][c]
    __bf16*   xs2  = (__bf16*)(smem + XS2_OFF);  // [c][pixel]
    float*    lg   = (float*)(smem + LG_OFF);
    __bf16*   al   = (__bf16*)(smem + AL_OFF);
    float*    red  = (float*)(smem + RED_OFF);
    float*    scl  = (float*)(smem + SC_OFF);

    const int tid   = threadIdx.x;
    const int lane  = tid & 31;
    const int wave  = tid >> 5;
    const int n     = blockIdx.x / PSPLIT;
    const int split = blockIdx.x % PSPLIT;
    const int p0wg  = split * PT;

    // ---- stage conv_w as bf16 into LDS (once per workgroup) ----
    {
        #pragma unroll 4
        for (int i = 0; i < (KK * CC) / 256; ++i) {
            int idx = tid + 256 * i;
            int k = idx >> 10, c = idx & 1023;
            cw[k * CW_STRIDE + c] = (__bf16)conv_w[idx];
        }
    }

    // vlad accumulators: wave owns clusters {0..31} x channels [wave*128, wave*128+128)
    v8f acc2[2][8];
    #pragma unroll
    for (int m = 0; m < 2; ++m)
        #pragma unroll
        for (int nt = 0; nt < 8; ++nt)
            acc2[m][nt] = 0.0f;

    float asum_reg = 0.0f;   // only meaningful for tid<32 (cluster = tid)

    __syncthreads();

    const int pl = lane;        // pixel within tile handled by this thread (loads)
    const int cg = wave;        // channel group 0..7

    for (int tile = 0; tile < PT / TILE_P; ++tile) {
        const int p0 = p0wg + tile * TILE_P;
        const float* xg = x + (size_t)n * CC * PP + p0 + pl;

        // ---- pass A: per-pixel sum of squares over C ----
        float ss = 0.0f;
        #pragma unroll 4
        for (int i = 0; i < CC / 8; ++i) {
            float v = xg[(size_t)(cg + 8 * i) * PP];
            ss += v * v;
        }
        red[cg * 32 + pl] = ss;
        __syncthreads();
        if (tid < 32) {
            float t = 0.0f;
            #pragma unroll
            for (int g = 0; g < 8; ++g) t += red[g * 32 + tid];
            scl[tid] = 1.0f / fmaxf(sqrtf(t), 1e-12f);
        }
        // zero logits accumulator
        #pragma unroll
        for (int i = 0; i < 4; ++i) lg[tid + 256 * i] = 0.0f;
        __syncthreads();

        // ---- pass B: normalize, convert to bf16, stage both LDS layouts ----
        {
            const float inv = scl[pl];
            #pragma unroll 4
            for (int i = 0; i < CC / 8; ++i) {
                int c = cg + 8 * i;
                __bf16 bv = (__bf16)(xg[(size_t)c * PP] * inv);
                xs1[pl * XS1_STRIDE + c] = bv;     // [pixel][c]
                xs2[c * XS2_STRIDE + pl] = bv;     // [c][pixel]
            }
        }
        __syncthreads();

        // ---- GEMM1: logits(32x32) = conv_w(32x1024) @ xnorm(1024x32) ----
        // wave -> (Mtile, Ntile, K-half); all operands are contiguous 2x b128 loads
        {
            const int Mtile = wave & 1;
            const int Ntile = (wave >> 1) & 1;
            const int kh    = wave >> 2;
            const int half  = lane >> 4;
            const int cluster = Mtile * 16 + (lane & 15);
            const int pix     = Ntile * 16 + (lane & 15);
            v8f acc1 = 0.0f;
            #pragma unroll
            for (int ks = 0; ks < 16; ++ks) {
                const int cbase = kh * 512 + ks * 32;
                // A: conv_w tile, ISA 16-bit A layout (k = g*16 + half*8 + j*2 (+1))
                union { uint4 q[2]; v16bf v; } A;
                const __bf16* ab = cw + cluster * CW_STRIDE + cbase + half * 8;
                A.q[0] = *(const uint4*)(ab);
                A.q[1] = *(const uint4*)(ab + 16);
                // B: same k order as A (k = g*16 + half*8 + j)
                union { uint4 q[2]; v16bf v; } B;
                const __bf16* bb = xs1 + pix * XS1_STRIDE + cbase + half * 8;
                B.q[0] = *(const uint4*)(bb);
                B.q[1] = *(const uint4*)(bb + 16);
                acc1 = __builtin_amdgcn_wmma_f32_16x16x32_bf16(
                           false, A.v, false, B.v, (short)0, acc1, false, false);
            }
            // combine K-halves into LDS logits
            #pragma unroll
            for (int r = 0; r < 8; ++r) {
                int m = r + (lane >> 4) * 8;
                atomicAdd(&lg[(Mtile * 16 + m) * TILE_P + Ntile * 16 + (lane & 15)], acc1[r]);
            }
        }
        __syncthreads();

        // ---- softmax over clusters (per pixel), then build A-matrix for GEMM2 ----
        if (tid < 32) {
            const int p = tid;
            float mx = -3.4e38f;
            for (int k = 0; k < KK; ++k) mx = fmaxf(mx, lg[k * TILE_P + p]);
            float s = 0.0f;
            for (int k = 0; k < KK; ++k) s += __expf(lg[k * TILE_P + p] - mx);
            float inv = 1.0f / s;
            for (int k = 0; k < KK; ++k)
                lg[k * TILE_P + p] = __expf(lg[k * TILE_P + p] - mx) * inv;
        }
        __syncthreads();
        if (tid < 32) {
            const int k = tid;
            float rs = 0.0f;
            for (int p = 0; p < TILE_P; ++p) {
                float av = lg[k * TILE_P + p];
                rs += av;
                al[k * AL_STRIDE + p] = (__bf16)av;
            }
            asum_reg += rs;
        }
        __syncthreads();

        // ---- GEMM2: vlad(32x1024) += a(32x32) @ xnorm^T(32x1024) ----
        {
            const int half = lane >> 4;
            // A operands (per M tile) shared across all 8 N tiles
            union { uint4 q[2]; v16bf v; } A0, A1;
            {
                const __bf16* ab0 = al + ((lane & 15)) * AL_STRIDE + half * 8;
                const __bf16* ab1 = al + (16 + (lane & 15)) * AL_STRIDE + half * 8;
                A0.q[0] = *(const uint4*)(ab0);  A0.q[1] = *(const uint4*)(ab0 + 16);
                A1.q[0] = *(const uint4*)(ab1);  A1.q[1] = *(const uint4*)(ab1 + 16);
            }
            #pragma unroll
            for (int nt = 0; nt < 8; ++nt) {
                const int c = wave * 128 + nt * 16 + (lane & 15);
                // B: pixels g*16 + half*8 + j at fixed c (same k order as A)
                union { uint4 q[2]; v16bf v; } B;
                const __bf16* bb = xs2 + c * XS2_STRIDE + half * 8;
                B.q[0] = *(const uint4*)(bb);
                B.q[1] = *(const uint4*)(bb + 16);
                acc2[0][nt] = __builtin_amdgcn_wmma_f32_16x16x32_bf16(
                                  false, A0.v, false, B.v, (short)0, acc2[0][nt], false, false);
                acc2[1][nt] = __builtin_amdgcn_wmma_f32_16x16x32_bf16(
                                  false, A1.v, false, B.v, (short)0, acc2[1][nt], false, false);
            }
        }
        __syncthreads();   // protect xs*/lg/al before next tile overwrites
    }

    // ---- flush partial vlad / asum to workspace ----
    float* vbase = vlad_ws + (size_t)n * KK * CC;
    #pragma unroll
    for (int m = 0; m < 2; ++m)
        #pragma unroll
        for (int nt = 0; nt < 8; ++nt)
            #pragma unroll
            for (int r = 0; r < 8; ++r) {
                int cluster = m * 16 + r + (lane >> 4) * 8;
                int c = wave * 128 + nt * 16 + (lane & 15);
                atomicAdd(&vbase[cluster * CC + c], acc2[m][nt][r]);
            }
    if (tid < 32) atomicAdd(&asum_ws[n * KK + tid], asum_reg);
}

// ---------------- finalize: centroid subtract + intra-cluster L2 norm ----------------
__global__ void __launch_bounds__(256)
netvlad_intra(const float* __restrict__ vlad_ws, const float* __restrict__ asum_ws,
              const float* __restrict__ cent, float* __restrict__ rowss,
              float* __restrict__ out) {
    __shared__ float red[256];
    const int nk = blockIdx.x;            // n*32 + k
    const int k  = nk & 31;
    const int tid = threadIdx.x;
    const float a = asum_ws[nk];
    float v[4];
    float ss = 0.0f;
    #pragma unroll
    for (int j = 0; j < 4; ++j) {
        int c = tid + 256 * j;
        v[j] = vlad_ws[(size_t)nk * CC + c] - a * cent[k * CC + c];
        ss += v[j] * v[j];
    }
    red[tid] = ss;
    __syncthreads();
    for (int s = 128; s > 0; s >>= 1) {
        if (tid < s) red[tid] += red[tid + s];
        __syncthreads();
    }
    const float tot = red[0];
    const float inv = 1.0f / fmaxf(sqrtf(tot), 1e-12f);
    if (tid == 0) rowss[nk] = tot * inv * inv;   // sumsq after intra-norm
    #pragma unroll
    for (int j = 0; j < 4; ++j)
        out[(size_t)nk * CC + tid + 256 * j] = v[j] * inv;
}

// ---------------- finalize: global L2 norm per image ----------------
__global__ void __launch_bounds__(256)
netvlad_global(const float* __restrict__ rowss, float* __restrict__ out) {
    __shared__ float gsc;
    const int n = blockIdx.x;
    const int tid = threadIdx.x;
    if (tid == 0) {
        float s = 0.0f;
        for (int k = 0; k < KK; ++k) s += rowss[n * KK + k];
        gsc = 1.0f / fmaxf(sqrtf(s), 1e-12f);
    }
    __syncthreads();
    const float g = gsc;
    float* ob = out + (size_t)n * KK * CC;
    #pragma unroll 4
    for (int j = 0; j < (KK * CC) / 256; ++j)
        ob[tid + 256 * j] *= g;
}

extern "C" void kernel_launch(void* const* d_in, const int* in_sizes, int n_in,
                              void* d_out, int out_size, void* d_ws, size_t ws_size,
                              hipStream_t stream) {
    const float* x      = (const float*)d_in[0];   // (16,1024,32,32)
    const float* conv_w = (const float*)d_in[1];   // (32,1024)
    const float* cent   = (const float*)d_in[2];   // (32,1024)
    float* out = (float*)d_out;                    // (16, 32768)
    float* wsf = (float*)d_ws;
    float* vlad_ws = wsf + WS_VLAD;
    float* asum_ws = wsf + WS_ASUM;
    float* rowss   = wsf + WS_ROWSS;

    netvlad_zero_ws<<<(WS_FLOATS + 255) / 256, 256, 0, stream>>>(wsf, WS_FLOATS);
    netvlad_main<<<NN * PSPLIT, 256, LDS_TOTAL, stream>>>(x, conv_w, vlad_ws, asum_ws);
    netvlad_intra<<<NN * KK, 256, 0, stream>>>(vlad_ws, asum_ws, cent, rowss, out);
    netvlad_global<<<NN, 256, 0, stream>>>(rowss, out);
}
